// SelfAttention_14559939133659
// MI455X (gfx1250) — matrix operand order repaired and checked by reading, not compiled, not run on hardware
//
#include <hip/hip_runtime.h>

#define B_  8
#define N_  1024
#define D_  1024
#define H_  16
#define HD_ 64
#define SCALE_ 0.125f  // 64^-0.5

typedef __bf16 bf16;
typedef __attribute__((ext_vector_type(16))) __bf16 bf16x16;
typedef __attribute__((ext_vector_type(8)))  __bf16 bf16x8;
typedef __attribute__((ext_vector_type(4)))  __bf16 bf16x4;
typedef __attribute__((ext_vector_type(8)))  float  f32x8;

static __device__ __forceinline__ bf16 f2bf(float f) {
    unsigned u = __builtin_bit_cast(unsigned, f);
    unsigned r = (u + 0x7FFFu + ((u >> 16) & 1u)) >> 16;   // round-to-nearest-even
    unsigned short s = (unsigned short)r;
    return __builtin_bit_cast(bf16, s);
}

static __device__ __forceinline__ bf16x16 cat8(bf16x8 lo, bf16x8 hi) {
    return __builtin_shufflevector(lo, hi, 0,1,2,3,4,5,6,7,8,9,10,11,12,13,14,15);
}

static __device__ __forceinline__ f32x8 wmma_bf16(bf16x16 a, bf16x16 b, f32x8 c) {
    // D = A(16x32 bf16) * B(32x16 bf16) + C(16x16 f32)
    return __builtin_amdgcn_wmma_f32_16x16x32_bf16(false, a, false, b,
                                                   (short)0, c, false, false);
}

// ---------------------------------------------------------------------------
// Async global->LDS copy (CDNA5 GLOBAL_LOAD_ASYNC_TO_LDS_B128, ASYNCcnt).
// Builtin signature (from compiler diagnostic): param0 = int4 addrspace(1)*.
// No data VGPRs: copies issue back-to-back, one wait before the barrier.
// ---------------------------------------------------------------------------
#if __has_builtin(__builtin_amdgcn_global_load_async_to_lds_b128)
#define ASYNC_LDS_COPY 1
typedef int v4i __attribute__((vector_size(16)));
typedef __attribute__((address_space(1))) v4i gv4i;   // global int4
typedef __attribute__((address_space(3))) v4i lv4i;   // LDS int4
static __device__ __forceinline__ void copy16_g2l(const bf16* g, bf16* l) {
    gv4i* gp = (gv4i*)(unsigned long long)(__SIZE_TYPE__)g;
    lv4i* lp = (lv4i*)(unsigned int)(__SIZE_TYPE__)l;   // low 32 bits = LDS offset
    __builtin_amdgcn_global_load_async_to_lds_b128(gp, lp, 0, 0);
}
static __device__ __forceinline__ void wait_lds_arrivals() {
    asm volatile("s_wait_asynccnt 0x0" ::: "memory");
}
#else
#define ASYNC_LDS_COPY 0
static __device__ __forceinline__ void copy16_g2l(const bf16* g, bf16* l) {
    *(bf16x8*)l = *(const bf16x8*)g;
}
static __device__ __forceinline__ void wait_lds_arrivals() {}
#endif

// ---------------------------------------------------------------------------
// fp32 -> bf16 conversion (grid-stride, 4-wide)
// ---------------------------------------------------------------------------
__global__ __launch_bounds__(256)
void cvt_f32_bf16(const float* __restrict__ src, bf16* __restrict__ dst, int n) {
    int stride = gridDim.x * blockDim.x * 4;
    for (int i = (blockIdx.x * blockDim.x + threadIdx.x) * 4; i < n; i += stride) {
        float4 v = *(const float4*)(src + i);
        bf16x4 o;
        o[0] = f2bf(v.x); o[1] = f2bf(v.y); o[2] = f2bf(v.z); o[3] = f2bf(v.w);
        *(bf16x4*)(dst + i) = o;
    }
}

// ===========================================================================
// Shared WMMA GEMM mainloop: C[128x64] = A[128xK] * W[64xK]^T, K-slab 64.
// 8 waves in 4(M) x 2(N); each wave owns a 32x32 tile (4 C fragments).
// Per slab per wave: 8 WMMAs, 16 ds_load_b128 (2.0 loads/WMMA).
// Slab fill via async LDS DMA (6 x b128 per thread, zero data VGPRs).
// ===========================================================================
struct GemmCtx {
    f32x8 acc[2][2];   // [cm][cn]
    int wm, wn, half, l16;
};

static __device__ __forceinline__ void gemm_mainloop(
        GemmCtx& g, bf16* As, bf16* Bs,
        const bf16* __restrict__ Ag, const bf16* __restrict__ Wg,
        int m0, int n0) {
    const int tid  = threadIdx.x;
    const int lane = tid & 31;
    g.half = lane >> 4;
    g.l16  = lane & 15;
    const int wave = tid >> 5;
    g.wm = wave >> 1;          // 0..3  -> rows wm*32
    g.wn = wave & 1;           // 0..1  -> cols wn*32
    g.acc[0][0] = (f32x8){}; g.acc[0][1] = (f32x8){};
    g.acc[1][0] = (f32x8){}; g.acc[1][1] = (f32x8){};

    const int kbB = g.half * 16;   // B-frag k-base within 32-chunk
    const int kbA = g.half * 16;   // A-frag k-base within 32-chunk

    for (int kk = 0; kk < D_; kk += 64) {
        // stage A slab: 128 rows x 64 k (16 KB) = 1024 x 16B chunks
        #pragma unroll
        for (int c = 0; c < 4; ++c) {
            int chunk = c * 256 + tid;
            int row = chunk >> 3, off = (chunk & 7) * 8;
            copy16_g2l(&Ag[(m0 + row) * D_ + kk + off], &As[row * 64 + off]);
        }
        // stage W slab: 64 rows x 64 k (8 KB) = 512 x 16B chunks
        #pragma unroll
        for (int c = 0; c < 2; ++c) {
            int chunk = c * 256 + tid;
            int row = chunk >> 3, off = (chunk & 7) * 8;
            copy16_g2l(&Wg[(n0 + row) * D_ + kk + off], &Bs[row * 64 + off]);
        }
        if (kk + 64 < D_) {   // WGP-scope prefetch of next slab
            __builtin_prefetch(&Ag[(m0 + (tid >> 1)) * D_ + kk + 64], 0, 3);
            __builtin_prefetch(&Wg[(n0 + (tid >> 2)) * D_ + kk + 64], 0, 3);
        }
        wait_lds_arrivals();
        __syncthreads();
        #pragma unroll
        for (int ks = 0; ks < 64; ks += 32) {
            bf16x16 bf[2], af[2];
            #pragma unroll
            for (int cn = 0; cn < 2; ++cn) {
                const int nl = g.wn * 32 + cn * 16 + g.l16;
                bf[cn] = *(const bf16x16*)&Bs[nl * 64 + ks + kbB];
            }
            #pragma unroll
            for (int cm = 0; cm < 2; ++cm) {
                const int ml = g.wm * 32 + cm * 16 + g.l16;
                bf16x8 lo = *(const bf16x8*)&As[ml * 64 + ks + kbA];
                bf16x8 hi = *(const bf16x8*)&As[ml * 64 + ks + kbA + 8];
                af[cm] = cat8(lo, hi);
            }
            #pragma unroll
            for (int cm = 0; cm < 2; ++cm)
                #pragma unroll
                for (int cn = 0; cn < 2; ++cn)
                    g.acc[cm][cn] = wmma_bf16(af[cm], bf[cn], g.acc[cm][cn]);
        }
        __syncthreads();
    }
}

// ---------------------------------------------------------------------------
// QKV projection: [8192,1024]x[1024,3072]^T + bias -> Q,K [b,h,n,hd], Vt [b,h,hd,n]
// ---------------------------------------------------------------------------
__global__ __launch_bounds__(256)
void qkv_gemm(const bf16* __restrict__ Xb, const bf16* __restrict__ Wb,
              const float* __restrict__ bias,
              bf16* __restrict__ Qb, bf16* __restrict__ Kb, bf16* __restrict__ Vtb) {
    __shared__ __attribute__((aligned(64))) bf16 As[128 * 64];
    __shared__ __attribute__((aligned(64))) bf16 Bs[64 * 64];
    const int n0 = blockIdx.x * 64;
    const int m0 = blockIdx.y * 128;

    GemmCtx g;
    gemm_mainloop(g, As, Bs, Xb, Wb, m0, n0);

    // block column range is exactly one (head, q/k/v) 64-wide section
    const int h     = blockIdx.x / 3;
    const int which = blockIdx.x % 3;     // 0=q 1=k 2=v
    #pragma unroll
    for (int cn = 0; cn < 2; ++cn) {
        const int hd = g.wn * 32 + cn * 16 + g.l16;
        const float bv = bias[n0 + hd];
        #pragma unroll
        for (int cm = 0; cm < 2; ++cm) {
            #pragma unroll
            for (int r = 0; r < 8; ++r) {
                int mg  = m0 + g.wm * 32 + cm * 16 + r + g.half * 8;
                int b   = mg >> 10, tok = mg & (N_ - 1);
                bf16 v  = f2bf(g.acc[cm][cn][r] + bv);
                if (which == 0)      Qb [((b * H_ + h) * N_  + tok) * HD_ + hd] = v;
                else if (which == 1) Kb [((b * H_ + h) * N_  + tok) * HD_ + hd] = v;
                else                 Vtb[((b * H_ + h) * HD_ + hd ) * N_ + tok] = v;
            }
        }
    }
}

// ---------------------------------------------------------------------------
// Output projection: [8192,1024]x[1024,1024]^T + fc_b + x  (fp32 out)
// ---------------------------------------------------------------------------
__global__ __launch_bounds__(256)
void fc_gemm(const bf16* __restrict__ Ab, const bf16* __restrict__ Wb,
             const float* __restrict__ bias, const float* __restrict__ xres,
             float* __restrict__ out) {
    __shared__ __attribute__((aligned(64))) bf16 As[128 * 64];
    __shared__ __attribute__((aligned(64))) bf16 Bs[64 * 64];
    const int n0 = blockIdx.x * 64;
    const int m0 = blockIdx.y * 128;

    GemmCtx g;
    gemm_mainloop(g, As, Bs, Ab, Wb, m0, n0);

    #pragma unroll
    for (int cn = 0; cn < 2; ++cn) {
        const int ng = n0 + g.wn * 32 + cn * 16 + g.l16;
        const float bv = bias[ng];
        #pragma unroll
        for (int cm = 0; cm < 2; ++cm) {
            #pragma unroll
            for (int r = 0; r < 8; ++r) {
                int mg = m0 + g.wm * 32 + cm * 16 + r + g.half * 8;
                out[mg * D_ + ng] = g.acc[cm][cn][r] + bv + xres[mg * D_ + ng];
            }
        }
    }
}

// ---------------------------------------------------------------------------
// Flash attention: block = (b,h, 128-query strip), 8 waves x 16 rows, 32-key tiles
// ---------------------------------------------------------------------------
__global__ __launch_bounds__(256)
void attn(const bf16* __restrict__ Qb, const bf16* __restrict__ Kb,
          const bf16* __restrict__ Vtb, bf16* __restrict__ Ob) {
    __shared__ __attribute__((aligned(64))) bf16 Ks[32 * 64];       // [key][hd]
    __shared__ __attribute__((aligned(64))) bf16 Vs[64 * 32];       // [hd][key]
    __shared__ __attribute__((aligned(64))) bf16 Ps[8 * 16 * 32];   // per-wave P tile
    const int tid  = threadIdx.x, lane = tid & 31, wave = tid >> 5;
    const int half = lane >> 4, l16 = lane & 15;
    const int bh    = blockIdx.x;
    const int qbase = blockIdx.y * 128 + wave * 16;

    // Q fragments (16 rows x 64 hd = 2 A-frags), kept in registers
    bf16x16 qf[2];
    {
        const bf16* qrow = Qb + (bh * N_ + qbase + l16) * HD_;
        const int kb = half * 16;
        #pragma unroll
        for (int f = 0; f < 2; ++f) {
            bf16x8 lo = *(const bf16x8*)(qrow + f * 32 + kb);
            bf16x8 hi = *(const bf16x8*)(qrow + f * 32 + kb + 8);
            qf[f] = cat8(lo, hi);
        }
    }

    f32x8 o[4] = {{}, {}, {}, {}};
    float mrun[8], lrun[8];
    #pragma unroll
    for (int r = 0; r < 8; ++r) { mrun[r] = -1e30f; lrun[r] = 0.f; }

    const int krow = tid >> 3, kchk = (tid & 7) * 8;
    const int vrow = tid >> 2, vchk = (tid & 3) * 8;
    bf16* Pw = Ps + wave * 512;

    for (int kt = 0; kt < N_; kt += 32) {
        copy16_g2l(&Kb [(bh * N_  + kt + krow) * HD_ + kchk], &Ks[krow * 64 + kchk]);
        copy16_g2l(&Vtb[(bh * HD_ + vrow) * N_ + kt  + vchk], &Vs[vrow * 32 + vchk]);
        wait_lds_arrivals();
        __syncthreads();

        // S = Q K^T  (16x32 in two 16x16 C tiles, contracted over hd=64)
        f32x8 s0 = {}, s1 = {};
        {
            const int kb = half * 16;
            #pragma unroll
            for (int f = 0; f < 2; ++f) {
                bf16x16 b0 = *(const bf16x16*)&Ks[l16 * 64        + f * 32 + kb];
                bf16x16 b1 = *(const bf16x16*)&Ks[(16 + l16) * 64 + f * 32 + kb];
                s0 = wmma_bf16(qf[f], b0, s0);
                s1 = wmma_bf16(qf[f], b1, s1);
            }
        }

        // online softmax; C-layout row r+8*half lives in this lane's element r
        #pragma unroll
        for (int r = 0; r < 8; ++r) {
            float a = s0[r] * SCALE_, c = s1[r] * SCALE_;
            float t = fmaxf(a, c);
            t = fmaxf(t, __shfl_xor(t, 1, 32));
            t = fmaxf(t, __shfl_xor(t, 2, 32));
            t = fmaxf(t, __shfl_xor(t, 4, 32));
            t = fmaxf(t, __shfl_xor(t, 8, 32));
            float mn = fmaxf(mrun[r], t);
            float al = __expf(mrun[r] - mn);
            float p0 = __expf(a - mn), p1 = __expf(c - mn);
            float rs = p0 + p1;
            rs += __shfl_xor(rs, 1, 32);
            rs += __shfl_xor(rs, 2, 32);
            rs += __shfl_xor(rs, 4, 32);
            rs += __shfl_xor(rs, 8, 32);
            lrun[r] = lrun[r] * al + rs;
            mrun[r] = mn;
            const int row = r + half * 8;
            Pw[row * 32 + l16]      = f2bf(p0);
            Pw[row * 32 + 16 + l16] = f2bf(p1);
            #pragma unroll
            for (int t4 = 0; t4 < 4; ++t4) o[t4][r] *= al;
        }
        asm volatile("s_wait_dscnt 0x0" ::: "memory");  // cross-lane LDS P visibility

        // re-load P as A-fragment (16x32), then O += P V
        bf16x16 pf;
        {
            const int kb = half * 16;
            bf16x8 lo = *(const bf16x8*)&Pw[l16 * 32 + kb];
            bf16x8 hi = *(const bf16x8*)&Pw[l16 * 32 + kb + 8];
            pf = cat8(lo, hi);
        }
        {
            const int kb = half * 16;
            #pragma unroll
            for (int t4 = 0; t4 < 4; ++t4) {
                bf16x16 vf = *(const bf16x16*)&Vs[(t4 * 16 + l16) * 32 + kb];
                o[t4] = wmma_bf16(pf, vf, o[t4]);
            }
        }
        __syncthreads();
    }

    // normalize + scatter to [b, n, h*64+hd] bf16
    const int b = bh >> 4, h = bh & 15;
    #pragma unroll
    for (int r = 0; r < 8; ++r) {
        int tok   = qbase + r + half * 8;
        float inv = 1.0f / lrun[r];
        bf16* orow = Ob + (b * N_ + tok) * D_ + h * HD_ + l16;
        #pragma unroll
        for (int t4 = 0; t4 < 4; ++t4) orow[t4 * 16] = f2bf(o[t4][r] * inv);
    }
}

// ---------------------------------------------------------------------------
extern "C" void kernel_launch(void* const* d_in, const int* in_sizes, int n_in,
                              void* d_out, int out_size, void* d_ws, size_t ws_size,
                              hipStream_t stream) {
    (void)in_sizes; (void)n_in; (void)out_size; (void)ws_size;
    const float* x     = (const float*)d_in[0];
    const float* qkv_w = (const float*)d_in[1];
    const float* qkv_b = (const float*)d_in[2];
    const float* fc_w  = (const float*)d_in[3];
    const float* fc_b  = (const float*)d_in[4];
    float* out = (float*)d_out;

    char* ws = (char*)d_ws;
    bf16* xbf  = (bf16*)(ws);                 // 16 MB
    bf16* qwbf = (bf16*)(ws + 16777216);      //  6 MB
    bf16* fwbf = (bf16*)(ws + 23068672);      //  2 MB
    bf16* Qb   = (bf16*)(ws + 25165824);      // 16 MB  [b,h,n,hd]
    bf16* Kb   = (bf16*)(ws + 41943040);      // 16 MB  [b,h,n,hd]
    bf16* Vtb  = (bf16*)(ws + 58720256);      // 16 MB  [b,h,hd,n]
    bf16* Ob   = (bf16*)(ws + 75497472);      // 16 MB  [b,n,d]

    cvt_f32_bf16<<<2048, 256, 0, stream>>>(x,     xbf,  B_ * N_ * D_);
    cvt_f32_bf16<<<2048, 256, 0, stream>>>(qkv_w, qwbf, 3 * D_ * D_);
    cvt_f32_bf16<<<1024, 256, 0, stream>>>(fc_w,  fwbf, D_ * D_);

    qkv_gemm<<<dim3(48, 64), 256, 0, stream>>>(xbf, qwbf, qkv_b, Qb, Kb, Vtb);
    attn    <<<dim3(128, 8), 256, 0, stream>>>(Qb, Kb, Vtb, Ob);
    fc_gemm <<<dim3(16, 64), 256, 0, stream>>>(Ob, fwbf, fc_b, x, out);
}
